// SimpleGNNEncoder_18837726560910
// MI455X (gfx1250) — matrix operand order repaired and checked by reading, not compiled, not run on hardware
//
#include <hip/hip_runtime.h>
#include <hip/hip_bf16.h>

typedef __attribute__((ext_vector_type(16))) _Float16 v16h;
typedef __attribute__((ext_vector_type(8)))  float    v8f;

union V16 { v16h v; unsigned u[8]; };

#define N_ATOMS   500000
#define N_GRAPHS  10000
#define HID       128
#define TILES_A   31250      // N_ATOMS / 16
#define TILES_B   625        // N_GRAPHS / 16
#define WAVES_A   4
#define WAVES_B   4

// ---------------------------------------------------------------------------
// Convert an fp32 weight matrix [Kreal x Ncols] (row-major) into fragment-major
// f16 B-matrix fragments. Block b = nt*KB + kb; 32 lanes each emit 16 halfs.
// ISA B layout (16-bit, 32x16): VGPR v<4 holds K=2v+8*hi,+1;
// v>=4 holds K=16+2(v-4)+8*hi,+1.  K >= Kreal padded with zeros.
// ---------------------------------------------------------------------------
__global__ void prep_wfrag(const float* __restrict__ w, _Float16* __restrict__ dst,
                           int Kreal, int Ncols, int KB) {
    int b    = blockIdx.x;        // nt*KB + kb
    int nt   = b / KB;
    int kb   = b % KB;
    int lane = threadIdx.x;       // 0..31
    int nl   = lane & 15;
    int hi   = lane >> 4;
    _Float16 frag[16];
#pragma unroll
    for (int v = 0; v < 8; ++v) {
        int K = kb * 32 + 2 * (v & 3) + ((v >= 4) ? 16 : 0) + 8 * hi;
        frag[2*v]   = (K     < Kreal) ? (_Float16)w[(size_t)K      * Ncols + nt*16 + nl] : (_Float16)0.f;
        frag[2*v+1] = (K + 1 < Kreal) ? (_Float16)w[(size_t)(K+1)  * Ncols + nt*16 + nl] : (_Float16)0.f;
    }
    _Float16* out = dst + ((size_t)b * 32 + lane) * 16;
#pragma unroll
    for (int i = 0; i < 16; ++i) out[i] = frag[i];
}

__global__ void count_kernel(const int* __restrict__ bidx, float* __restrict__ counts, int n) {
    int i = blockIdx.x * blockDim.x + threadIdx.x;
    if (i < n) atomicAdd(&counts[bidx[i]], 1.0f);
}

__device__ __forceinline__ float silu(float x) {
    return x / (1.f + __expf(-x));
}

// ---------------------------------------------------------------------------
// Node MLP: h = silu([emb(species);fracs] @ w1 + b1) @ w2 + b2,
// then run-length-compressed atomic scatter-add into pooled[graph][128].
// One wave handles 16 atoms.
// ---------------------------------------------------------------------------
__global__ __launch_bounds__(WAVES_A * 32)
void node_mlp_kernel(const float* __restrict__ fracs, const int* __restrict__ species,
                     const int* __restrict__ bidx, const float* __restrict__ emb,
                     const _Float16* __restrict__ w1f, const float* __restrict__ b1,
                     const _Float16* __restrict__ w2f, const float* __restrict__ b2,
                     float* __restrict__ pooled) {
    // padded strides: 100 halfs (200B) and 132 halfs (264B) -> bank-conflict free
    __shared__ _Float16 xs[WAVES_A][16][100];
    __shared__ _Float16 hs[WAVES_A][16][132];

    int wave = threadIdx.x >> 5;
    int lane = threadIdx.x & 31;
    int tile = blockIdx.x * WAVES_A + wave;
    if (tile >= TILES_A) return;               // wave-uniform: EXEC stays all-1s
    int base = tile * 16;

    // --- stage x tile [16 x 96] as f16 in LDS ---
    for (int mr = 0; mr < 16; ++mr) {
        int a  = base + mr;
        int sp = species[a];                   // wave-uniform -> scalar load
        for (int f = lane; f < 96; f += 32) {
            float v;
            if (f < 64)      v = emb[sp * 64 + f];
            else if (f < 67) v = fracs[a * 3 + (f - 64)];
            else             v = 0.f;
            xs[wave][mr][f] = (_Float16)v;
        }
    }

    int m  = lane & 15;
    int hi = lane >> 4;

    // --- A fragments for layer 1 (K padded 67 -> 96, 3 blocks of 32) ---
    // ISA A layout (16-bit, 16x32): v<4 -> K=2v+8*hi ; v>=4 -> K=16+2(v-4)+8*hi
    V16 a1[3];
#pragma unroll
    for (int kb = 0; kb < 3; ++kb)
#pragma unroll
        for (int v = 0; v < 8; ++v) {
            int kk = kb * 32 + 2 * (v & 3) + ((v >= 4) ? 16 : 0) + 8 * hi;
            a1[kb].u[v] = *(const unsigned*)&xs[wave][m][kk];
        }

    const v16h* w1v = (const v16h*)w1f;
#pragma unroll
    for (int nt = 0; nt < 8; ++nt) {
        int n = nt * 16 + m;
        float bias = b1[n];
        v8f acc = {bias, bias, bias, bias, bias, bias, bias, bias};
#pragma unroll
        for (int kb = 0; kb < 3; ++kb) {
            v16h bf = w1v[(nt * 3 + kb) * 32 + lane];
            acc = __builtin_amdgcn_wmma_f32_16x16x32_f16(false, a1[kb].v, false, bf,
                                                         (short)0, acc, false, false);
        }
#pragma unroll
        for (int r = 0; r < 8; ++r)
            hs[wave][r + 8 * hi][n] = (_Float16)silu(acc[r]);
    }

    // --- A fragments for layer 2 (K = 128, 4 blocks of 32) ---
    V16 a2[4];
#pragma unroll
    for (int kb = 0; kb < 4; ++kb)
#pragma unroll
        for (int v = 0; v < 8; ++v) {
            int kk = kb * 32 + 2 * (v & 3) + ((v >= 4) ? 16 : 0) + 8 * hi;
            a2[kb].u[v] = *(const unsigned*)&hs[wave][m][kk];
        }

    // graph ids for this lane's 8 C rows (M = r + 8*hi)
    int g[8];
#pragma unroll
    for (int r = 0; r < 8; ++r) g[r] = bidx[base + 8 * hi + r];

    const v16h* w2v = (const v16h*)w2f;
#pragma unroll
    for (int nt = 0; nt < 8; ++nt) {
        int n = nt * 16 + m;
        float bias = b2[n];
        v8f acc = {bias, bias, bias, bias, bias, bias, bias, bias};
#pragma unroll
        for (int kb = 0; kb < 4; ++kb) {
            v16h bf = w2v[(nt * 4 + kb) * 32 + lane];
            acc = __builtin_amdgcn_wmma_f32_16x16x32_f16(false, a2[kb].v, false, bf,
                                                         (short)0, acc, false, false);
        }
        // run-length compressed scatter (batch indices are sorted)
        float run = acc[0];
        int   cg  = g[0];
#pragma unroll
        for (int r = 1; r < 8; ++r) {
            if (g[r] == cg) run += acc[r];
            else {
                atomicAdd(&pooled[(size_t)cg * HID + n], run);
                cg = g[r]; run = acc[r];
            }
        }
        atomicAdd(&pooled[(size_t)cg * HID + n], run);
    }
}

// ---------------------------------------------------------------------------
// Graph head: out = silu([pooled/count ; lattice] @ w3 + b3) @ w4 + b4,
// split into mu / logvar. One wave handles 16 graphs.
// ---------------------------------------------------------------------------
__global__ __launch_bounds__(WAVES_B * 32)
void graph_head_kernel(const float* __restrict__ pooled, const float* __restrict__ counts,
                       const float* __restrict__ lattice,
                       const _Float16* __restrict__ w3f, const float* __restrict__ b3,
                       const _Float16* __restrict__ w4f, const float* __restrict__ b4,
                       float* __restrict__ out) {
    __shared__ _Float16 gs[WAVES_B][16][164];   // 164 halfs = 328B stride
    __shared__ _Float16 hs[WAVES_B][16][132];

    int wave = threadIdx.x >> 5;
    int lane = threadIdx.x & 31;
    int tile = blockIdx.x * WAVES_B + wave;
    if (tile >= TILES_B) return;
    int base = tile * 16;

    // --- stage g tile [16 x 160] (128 pooled-mean + 9 lattice + pad) ---
    for (int mr = 0; mr < 16; ++mr) {
        int gph = base + mr;
        float inv = 1.f / counts[gph];          // every graph has >= 1 atom
        for (int f = lane; f < 160; f += 32) {
            float v;
            if (f < 128)      v = pooled[(size_t)gph * HID + f] * inv;
            else if (f < 137) v = lattice[gph * 9 + (f - 128)];
            else              v = 0.f;
            gs[wave][mr][f] = (_Float16)v;
        }
    }

    int m  = lane & 15;
    int hi = lane >> 4;

    // --- layer 3: K padded 137 -> 160 (5 blocks) ---
    V16 a3[5];
#pragma unroll
    for (int kb = 0; kb < 5; ++kb)
#pragma unroll
        for (int v = 0; v < 8; ++v) {
            int kk = kb * 32 + 2 * (v & 3) + ((v >= 4) ? 16 : 0) + 8 * hi;
            a3[kb].u[v] = *(const unsigned*)&gs[wave][m][kk];
        }

    const v16h* w3v = (const v16h*)w3f;
#pragma unroll
    for (int nt = 0; nt < 8; ++nt) {
        int n = nt * 16 + m;
        float bias = b3[n];
        v8f acc = {bias, bias, bias, bias, bias, bias, bias, bias};
#pragma unroll
        for (int kb = 0; kb < 5; ++kb) {
            v16h bf = w3v[(nt * 5 + kb) * 32 + lane];
            acc = __builtin_amdgcn_wmma_f32_16x16x32_f16(false, a3[kb].v, false, bf,
                                                         (short)0, acc, false, false);
        }
#pragma unroll
        for (int r = 0; r < 8; ++r)
            hs[wave][r + 8 * hi][n] = (_Float16)silu(acc[r]);
    }

    // --- layer 4: K = 128 (4 blocks), N = 256 (16 tiles) ---
    V16 a4[4];
#pragma unroll
    for (int kb = 0; kb < 4; ++kb)
#pragma unroll
        for (int v = 0; v < 8; ++v) {
            int kk = kb * 32 + 2 * (v & 3) + ((v >= 4) ? 16 : 0) + 8 * hi;
            a4[kb].u[v] = *(const unsigned*)&hs[wave][m][kk];
        }

    const v16h* w4v = (const v16h*)w4f;
#pragma unroll
    for (int nt = 0; nt < 16; ++nt) {
        int n = nt * 16 + m;
        float bias = b4[n];
        v8f acc = {bias, bias, bias, bias, bias, bias, bias, bias};
#pragma unroll
        for (int kb = 0; kb < 4; ++kb) {
            v16h bf = w4v[(nt * 4 + kb) * 32 + lane];
            acc = __builtin_amdgcn_wmma_f32_16x16x32_f16(false, a4[kb].v, false, bf,
                                                         (short)0, acc, false, false);
        }
#pragma unroll
        for (int r = 0; r < 8; ++r) {
            int gph = base + r + 8 * hi;
            float val = acc[r];
            if (n < 128) out[(size_t)gph * 128 + n] = val;                       // mu
            else         out[(size_t)N_GRAPHS * 128 + (size_t)gph * 128 + (n - 128)] = val; // logvar
        }
    }
}

// ---------------------------------------------------------------------------
extern "C" void kernel_launch(void* const* d_in, const int* in_sizes, int n_in,
                              void* d_out, int out_size, void* d_ws, size_t ws_size,
                              hipStream_t stream) {
    const float* lattice = (const float*)d_in[0];
    const float* fracs   = (const float*)d_in[1];
    const int*   species = (const int*)  d_in[2];
    const int*   bidx    = (const int*)  d_in[3];
    // d_in[4] (num_atoms_list) unused by reference
    const float* emb = (const float*)d_in[5];
    const float* w1  = (const float*)d_in[6];  const float* b1 = (const float*)d_in[7];
    const float* w2  = (const float*)d_in[8];  const float* b2 = (const float*)d_in[9];
    const float* w3  = (const float*)d_in[10]; const float* b3 = (const float*)d_in[11];
    const float* w4  = (const float*)d_in[12]; const float* b4 = (const float*)d_in[13];
    float* out = (float*)d_out;

    char* ws = (char*)d_ws;
    float* pooled = (float*)ws;                                 // 10000*128*4 = 5,120,000 B
    float* counts = (float*)(ws + 5120000);                     //    10000*4 =    40,000 B
    _Float16* w1f = (_Float16*)(ws + 5160192);                  // 3*8  frags = 24,576 B
    _Float16* w2f = w1f + 3 * 8 * 32 * 16;                      // 4*8  frags = 32,768 B
    _Float16* w3f = w2f + 4 * 8 * 32 * 16;                      // 5*8  frags = 40,960 B
    _Float16* w4f = w3f + 5 * 8 * 32 * 16;                      // 4*16 frags = 65,536 B

    // zero pooled + counts (contiguous)
    hipMemsetAsync(ws, 0, 5160000, stream);

    // weight fragment prep (fp32 -> f16 fragment-major, stays in L2)
    prep_wfrag<<< 8 * 3, 32, 0, stream>>>(w1, w1f,  67, 128, 3);
    prep_wfrag<<< 8 * 4, 32, 0, stream>>>(w2, w2f, 128, 128, 4);
    prep_wfrag<<< 8 * 5, 32, 0, stream>>>(w3, w3f, 137, 128, 5);
    prep_wfrag<<<16 * 4, 32, 0, stream>>>(w4, w4f, 128, 256, 4);

    // segment counts
    count_kernel<<<(N_ATOMS + 255) / 256, 256, 0, stream>>>(bidx, counts, N_ATOMS);

    // node MLP + pooled scatter
    node_mlp_kernel<<<(TILES_A + WAVES_A - 1) / WAVES_A, WAVES_A * 32, 0, stream>>>(
        fracs, species, bidx, emb, w1f, b1, w2f, b2, pooled);

    // graph head -> mu / logvar
    graph_head_kernel<<<(TILES_B + WAVES_B - 1) / WAVES_B, WAVES_B * 32, 0, stream>>>(
        pooled, counts, lattice, w3f, b3, w4f, b4, out);
}
